// InteractingLayer_6004364280328
// MI455X (gfx1250) — matrix operand order repaired and checked by reading, not compiled, not run on hardware
//
#include <hip/hip_runtime.h>

typedef __attribute__((ext_vector_type(16))) __bf16 v16bf;
typedef __attribute__((ext_vector_type(2)))  __bf16 v2bf;
typedef __attribute__((ext_vector_type(8)))  float  v8f;
typedef __attribute__((ext_vector_type(4)))  unsigned int v4u;
typedef __attribute__((ext_vector_type(8)))  int v8i;
typedef __attribute__((ext_vector_type(4)))  int v4i;

#define NB 16384   // batch
#define NF 64      // fields
#define NE 128     // embedding
#define ND 32      // per-head dim
#define NH 4       // heads
#define LDE 128    // row stride (elements) of x / q / k / v / out

// LDS layout (bytes). The first 64 KB region is time-multiplexed:
//   phase 0: fp32 x staging (32 KB)
//   phase 2/3: f32 scores [4][64][64] (64 KB)
//   phase 4: f32 fused output tile (32 KB) for async store-from-LDS
#define OFF_S   0
#define OFF_X   65536    // bf16 x      [64][128]    = 16 KB
#define OFF_Q   81920    // bf16 q      [64][128]    = 16 KB
#define OFF_K   98304    // bf16 k      [64][128]    = 16 KB
#define OFF_V   114688   // bf16 v      [64][128]    = 16 KB
#define OFF_RES 131072   // f32 res     [64][128]    = 32 KB
#define OFF_P   163840   // bf16 probs  [4][64][64]  = 32 KB
#define SMEM_BYTES (192 * 1024)

// Packed-weight fragment layout in d_ws: frag = mat*32 + kt*8 + nt, each
// fragment is 512 bf16 (one 32x16 B tile) stored lane-major: lane*16..lane*16+15.
#define WPACK_FRAGS 128

// ---- WMMA fragment helpers (wave32; A/B use contiguous k = 16*(lane>>4) + i) ----------

// A fragment: 16x32 bf16 (MxK). lane&15 = row M; lane>=16 selects K 16..31.
static __device__ inline v16bf load_frag_a(const __bf16* base, int ld,
                                           int row0, int k0, int lane) {
  const __bf16* p = base + (row0 + (lane & 15)) * ld + k0 + ((lane >> 4) << 4);
  v16bf a;
#pragma unroll
  for (int i = 0; i < 8; ++i) a[i] = p[i];
#pragma unroll
  for (int i = 0; i < 8; ++i) a[8 + i] = p[8 + i];
  return a;
}

// B fragment: 32x16 bf16 (KxN), row-major storage (stride ld).
static __device__ inline v16bf load_frag_b(const __bf16* base, int ld,
                                           int k0, int col0, int lane) {
  int col = col0 + (lane & 15);
  int kb = k0 + ((lane >> 4) << 4);
  v16bf b;
#pragma unroll
  for (int i = 0; i < 16; ++i) b[i] = base[(kb + i) * ld + col];
  return b;
}

// B fragment from transposed storage: B[k][n] = base[(col0+n)*ld + coff + k].
static __device__ inline v16bf load_frag_bT(const __bf16* base, int ld,
                                            int k0, int col0, int coff, int lane) {
  const __bf16* p = base + (col0 + (lane & 15)) * ld + coff + k0 + ((lane >> 4) << 4);
  v16bf b;
#pragma unroll
  for (int i = 0; i < 16; ++i) b[i] = p[i];
  return b;
}

// B fragment from the pre-packed weight buffer: one 32 B contiguous read/lane.
static __device__ inline v16bf load_frag_b_packed(const __bf16* wp, int frag, int lane) {
  const __bf16* p = wp + frag * 512 + lane * 16;
  v16bf b;
#pragma unroll
  for (int i = 0; i < 16; ++i) b[i] = p[i];
  return b;
}

// C/D fragment store: VGPR r holds row (lane>=16 ? 8 : 0) + r, col = lane&15.
static __device__ inline void store_c_bf16(__bf16* base, int ld, int row0, int col0,
                                           int lane, v8f c) {
  int col = col0 + (lane & 15);
  int r0 = row0 + ((lane >> 4) << 3);
#pragma unroll
  for (int r = 0; r < 8; ++r) base[(r0 + r) * ld + col] = (__bf16)c[r];
}

static __device__ inline void store_c_f32(float* base, int ld, int row0, int col0,
                                          int lane, v8f c) {
  int col = col0 + (lane & 15);
  int r0 = row0 + ((lane >> 4) << 3);
#pragma unroll
  for (int r = 0; r < 8; ++r) base[(r0 + r) * ld + col] = c[r];
}

#define WMMA_BF16(a, b, c) \
  __builtin_amdgcn_wmma_f32_16x16x32_bf16(false, (a), false, (b), (short)0, (c), false, false)

// ---------------------------------------------------------------------------
// One-time weight repack: fp32 (128x128, row-major) -> bf16 fragment-major.
__global__ __launch_bounds__(32) void pack_weights_kernel(
    const float* __restrict__ Wq, const float* __restrict__ Wk,
    const float* __restrict__ Wv, const float* __restrict__ Wr,
    __bf16* __restrict__ wpack) {
  int f = blockIdx.x;        // frag = mat*32 + kt*8 + nt
  int lane = threadIdx.x;
  int mat = f >> 5;
  int kt = (f >> 3) & 3;
  int nt = f & 7;
  const float* w = (mat == 0) ? Wq : (mat == 1) ? Wk : (mat == 2) ? Wv : Wr;
  int col = nt * 16 + (lane & 15);
  int kb = kt * 32 + ((lane >> 4) << 4);
  __bf16* dst = wpack + f * 512 + lane * 16;
#pragma unroll
  for (int i = 0; i < 16; ++i) dst[i] = (__bf16)w[(kb + i) * 128 + col];
}

// ---------------------------------------------------------------------------

__global__ __launch_bounds__(256) void autoint_interacting_kernel(
    const float* __restrict__ x, const __bf16* __restrict__ wpack,
    float* __restrict__ out) {
  extern __shared__ char smem[];
  float* sXf = (float*)(smem + OFF_S);   // fp32 staging (phase 0)
  float* sS = (float*)(smem + OFF_S);    // f32 scores [h][64][64] (phases 2-3)
  float* sOut = (float*)(smem + OFF_S);  // f32 output tile (phase 4)
  __bf16* sX = (__bf16*)(smem + OFF_X);
  __bf16* sQ = (__bf16*)(smem + OFF_Q);
  __bf16* sK = (__bf16*)(smem + OFF_K);
  __bf16* sV = (__bf16*)(smem + OFF_V);
  float* sRes = (float*)(smem + OFF_RES);
  __bf16* sP = (__bf16*)(smem + OFF_P);  // bf16 probs [h][64][64]

  const int tid = threadIdx.x;
  const int lane = tid & 31;
  // Wave id as a true scalar so all wave-level control flow is s_cbranch.
  const int waveU = __builtin_amdgcn_readfirstlane(tid >> 5);
  const float* xb = x + (size_t)blockIdx.x * (NF * LDE);

  __builtin_prefetch(wpack, 0, 3);  // packed weights stay hot in GL2

  // ---- Phase 0: TDM load of the 64x128 fp32 x tile into LDS ---------------
#if __has_builtin(__builtin_amdgcn_tensor_load_to_lds)
  if (waveU == 0) {
    unsigned ldsAddr = (unsigned)(size_t)(void*)sXf;
    unsigned long long ga = (unsigned long long)(size_t)xb;
    // D# group 0: count=1, lds_addr, global_addr[56:0], type=2 ("image")
    v4u g0;
    g0[0] = 1u;
    g0[1] = ldsAddr;
    g0[2] = (unsigned)(ga & 0xffffffffu);
    g0[3] = (unsigned)((ga >> 32) & 0x01ffffffu) | (2u << 30);
    // D# group 1: data_size=4B; tensor 128x64; tile 128x64; dim0 stride 128
    v8i g1;
    g1[0] = (int)(2u << 16);          // data_size = 2 (4 bytes)
    g1[1] = (int)(128u << 16);        // tensor_dim0[15:0] = 128
    g1[2] = (int)(64u << 16);         // tensor_dim0 hi | tensor_dim1 lo = 64
    g1[3] = (int)(128u << 16);        // tensor_dim1 hi | tile_dim0 = 128
    g1[4] = 64;                       // tile_dim1 = 64, tile_dim2 = 0
    g1[5] = 128;                      // tensor_dim0_stride lo = 128
    g1[6] = (int)(0x2000u << 16);     // dim0_stride hi | dim1_stride lo = 8192
    g1[7] = 0;
    v4i gz = {};
#if __clang_major__ >= 23
    v8i gz8 = {};
    __builtin_amdgcn_tensor_load_to_lds(g0, g1, gz, gz, gz8, 0);
#else
    __builtin_amdgcn_tensor_load_to_lds(g0, g1, gz, gz, 0);
#endif
    __builtin_amdgcn_s_wait_tensorcnt(0);
  }
#else
  {
    unsigned ldsBase = (unsigned)(size_t)(void*)sXf;
#pragma unroll
    for (int pass = 0; pass < 8; ++pass) {
      unsigned ldsAddr = ldsBase + (unsigned)(pass * 4096 + tid * 16);
      unsigned long long gaddr =
          (unsigned long long)(size_t)xb + (unsigned long long)(pass * 4096 + tid * 16);
      asm volatile("global_load_async_to_lds_b128 %0, %1, off"
                   :: "v"(ldsAddr), "v"(gaddr) : "memory");
    }
    asm volatile("s_wait_asynccnt 0" ::: "memory");
  }
#endif
  __syncthreads();

  // fp32 -> bf16 (8192 elements, 32 per thread; lowers to packed cvt)
#pragma unroll
  for (int i = 0; i < 32; ++i) {
    int idx = tid + i * 256;
    sX[idx] = (__bf16)sXf[idx];
  }
  __syncthreads();

  // ---- Phase 1: projections Q,K,V,Res = X(64x128) @ W(128x128) ------------
  // Wave pair (waveU>>1) owns one weight matrix; half (waveU&1) owns 2 row
  // blocks. A-fragments loaded once per row block, reused across 8 N-tiles.
  {
    const int matU = waveU >> 1;  // 0..3 (scalar)
    const int half = waveU & 1;   // 0..1 (scalar)
    __bf16* dstQKV = (matU == 0) ? sQ : (matU == 1) ? sK : sV;
#pragma unroll
    for (int rr = 0; rr < 2; ++rr) {
      int row0 = (half * 2 + rr) << 4;
      v16bf a0 = load_frag_a(sX, LDE, row0, 0, lane);
      v16bf a1 = load_frag_a(sX, LDE, row0, 32, lane);
      v16bf a2 = load_frag_a(sX, LDE, row0, 64, lane);
      v16bf a3 = load_frag_a(sX, LDE, row0, 96, lane);
      if (matU < 3) {
        for (int nt = 0; nt < 8; ++nt) {
          v8f acc = {};
          acc = WMMA_BF16(a0, load_frag_b_packed(wpack, matU * 32 + nt, lane), acc);
          acc = WMMA_BF16(a1, load_frag_b_packed(wpack, matU * 32 + 8 + nt, lane), acc);
          acc = WMMA_BF16(a2, load_frag_b_packed(wpack, matU * 32 + 16 + nt, lane), acc);
          acc = WMMA_BF16(a3, load_frag_b_packed(wpack, matU * 32 + 24 + nt, lane), acc);
          store_c_bf16(dstQKV, LDE, row0, nt << 4, lane, acc);
        }
      } else {
        for (int nt = 0; nt < 8; ++nt) {
          v8f acc = {};
          acc = WMMA_BF16(a0, load_frag_b_packed(wpack, 96 + nt, lane), acc);
          acc = WMMA_BF16(a1, load_frag_b_packed(wpack, 104 + nt, lane), acc);
          acc = WMMA_BF16(a2, load_frag_b_packed(wpack, 112 + nt, lane), acc);
          acc = WMMA_BF16(a3, load_frag_b_packed(wpack, 120 + nt, lane), acc);
          store_c_f32(sRes, LDE, row0, nt << 4, lane, acc);
        }
      }
    }
  }
  __syncthreads();

  // ---- Phase 2: scores[h] = Q_h(64x32) @ K_hT(32x64); K=32 in one WMMA ----
  // Wave pair owns one head; A-fragment reused across the 4 key tiles.
  {
    const int h = waveU >> 1;
    const int half = waveU & 1;
#pragma unroll
    for (int qq = 0; qq < 2; ++qq) {
      int qi = (half * 2 + qq) << 4;
      v16bf a = load_frag_a(sQ, LDE, qi, h * ND, lane);
#pragma unroll
      for (int kj = 0; kj < NF; kj += 16) {
        v16bf b = load_frag_bT(sK, LDE, 0, kj, h * ND, lane);
        v8f acc = {};
        acc = WMMA_BF16(a, b, acc);
        store_c_f32(sS + h * 4096, NF, qi, kj, lane, acc);
      }
    }
  }
  __syncthreads();

  // ---- Phase 3: softmax over each score row (4 heads x 64 rows = 256) -----
  {
    int h = tid >> 6;
    int r = tid & 63;
    float4* srow4 = (float4*)(sS + h * 4096 + r * NF);
    float4 buf[16];
    float m = -3.4e38f;
#pragma unroll
    for (int j = 0; j < 16; ++j) {
      buf[j] = srow4[j];
      m = fmaxf(m, fmaxf(fmaxf(buf[j].x, buf[j].y), fmaxf(buf[j].z, buf[j].w)));
    }
    float sum = 0.f;
#pragma unroll
    for (int j = 0; j < 16; ++j) {
      buf[j].x = __expf(buf[j].x - m);
      buf[j].y = __expf(buf[j].y - m);
      buf[j].z = __expf(buf[j].z - m);
      buf[j].w = __expf(buf[j].w - m);
      sum += buf[j].x + buf[j].y + buf[j].z + buf[j].w;
    }
    float inv = 1.0f / sum;
    v2bf* prow2 = (v2bf*)(sP + h * 4096 + r * NF);
#pragma unroll
    for (int j = 0; j < 16; ++j) {
      v2bf p0, p1;
      p0[0] = (__bf16)(buf[j].x * inv);
      p0[1] = (__bf16)(buf[j].y * inv);
      p1[0] = (__bf16)(buf[j].z * inv);
      p1[1] = (__bf16)(buf[j].w * inv);
      prow2[2 * j] = p0;
      prow2[2 * j + 1] = p1;
    }
  }
  __syncthreads();

  // ---- Phase 4: out_h = P_h(64x64) @ V_h(64x32); + res, ReLU -> LDS -------
  // (sS is dead after phase 3; its region now holds the f32 output tile.)
  {
    const int h = waveU >> 1;
    const int half = waveU & 1;
#pragma unroll
    for (int qq = 0; qq < 2; ++qq) {
      int qi = (half * 2 + qq) << 4;
      v16bf a0 = load_frag_a(sP + h * 4096, NF, qi, 0, lane);
      v16bf a1 = load_frag_a(sP + h * 4096, NF, qi, 32, lane);
#pragma unroll
      for (int dj = 0; dj < 2; ++dj) {
        int col0 = h * ND + (dj << 4);
        v8f acc = {};
        acc = WMMA_BF16(a0, load_frag_b(sV, LDE, 0, col0, lane), acc);
        acc = WMMA_BF16(a1, load_frag_b(sV, LDE, 32, col0, lane), acc);
        int col = col0 + (lane & 15);
        int r0 = qi + ((lane >> 4) << 3);
#pragma unroll
        for (int r = 0; r < 8; ++r) {
          float v = acc[r] + sRes[(r0 + r) * LDE + col];
          sOut[(r0 + r) * LDE + col] = v > 0.f ? v : 0.f;
        }
      }
    }
  }
  __syncthreads();

  // ---- Phase 5: fire-and-forget async store of the 32 KB output tile ------
  // S_ENDPGM performs an implicit wait-idle, draining ASYNCcnt at wave exit.
  {
    unsigned ldsBase = (unsigned)(size_t)(void*)sOut;
    unsigned long long gbase =
        (unsigned long long)(size_t)(out + (size_t)blockIdx.x * (NF * LDE));
#pragma unroll
    for (int pass = 0; pass < 8; ++pass) {
      unsigned off = (unsigned)(pass * 4096 + tid * 16);
      unsigned ldsAddr = ldsBase + off;
      unsigned long long ga = gbase + (unsigned long long)off;
      asm volatile("global_store_async_from_lds_b128 %0, %1, off"
                   :: "v"(ga), "v"(ldsAddr) : "memory");
    }
  }
}

extern "C" void kernel_launch(void* const* d_in, const int* in_sizes, int n_in,
                              void* d_out, int out_size, void* d_ws, size_t ws_size,
                              hipStream_t stream) {
  const float* x = (const float*)d_in[0];
  const float* Wq = (const float*)d_in[1];
  const float* Wk = (const float*)d_in[2];
  const float* Wv = (const float*)d_in[3];
  const float* Wr = (const float*)d_in[4];
  float* out = (float*)d_out;
  __bf16* wpack = (__bf16*)d_ws;  // 128 fragments x 512 bf16 = 128 KB

  pack_weights_kernel<<<dim3(WPACK_FRAGS), dim3(32), 0, stream>>>(Wq, Wk, Wv, Wr, wpack);
  autoint_interacting_kernel<<<dim3(NB), dim3(256), SMEM_BYTES, stream>>>(x, wpack, out);
}
